// OptimizedProtoHyperFormer_75007308857856
// MI455X (gfx1250) — compile-verified, matrix operand order checked by reading, not run on hardware
//
#include <hip/hip_runtime.h>
#include <hip/hip_bf16.h>
#include <math.h>

// ---------------- model dims ----------------
#define BB    32
#define IBAND 103
#define HDIM  256
#define WDIM  256
#define PP    16
#define RBC   8          // reduced bands
#define DMODEL 192
#define NTOK  256        // tokens per batch (16x16 patches)
#define NROWS (BB*NTOK)  // 8192 token rows
#define KS    24
#define FFH   384
#define NC    9
#define TAU   (0.07f + 1e-8f)

typedef __attribute__((ext_vector_type(16))) __bf16 v16bf;
typedef __attribute__((ext_vector_type(8)))  __bf16 v8bf;
typedef __attribute__((ext_vector_type(8)))  float  v8f;

// gemm output modes
#define OUT_F32       0
#define OUT_F32_RESID 1
#define OUT_BF_RM     2
#define OUT_BF_PP     3   // group-of-8 K-packed along M (for use as next B matrix)
#define OUT_BF_TR     4   // transposed bf16 (for keys^T)

// ============================================================
// Kernel 1: fused band-reduction + depthwise patch conv + pointwise
// grid = (BB * 16) blocks, one per (batch, patch-row). 256 threads = columns.
// Reads x (864 MB) exactly once, fully coalesced.  -> tok_raw (B,N,192) f32
// ============================================================
__global__ __launch_bounds__(256) void k_patch_embed(
    const float* __restrict__ x, const float* __restrict__ band_w,
    const float* __restrict__ band_b, const float* __restrict__ dw_w,
    const float* __restrict__ pw_w, float* __restrict__ tok_raw)
{
  __shared__ float s_dwq[PP*PP*RBC];   // [pq][c]
  __shared__ float s_band[RBC*IBAND];  // [c][ib]
  __shared__ float s_pw[DMODEL*RBC];   // [d][c]
  __shared__ float s_red[256*RBC];
  __shared__ float s_dw[16*RBC];
  __shared__ float s_sumdw[RBC];

  const int tid = threadIdx.x;
  for (int idx = tid; idx < RBC*PP*PP; idx += 256) {
    int c = idx >> 8; int pq = idx & 255;
    s_dwq[pq*RBC + c] = dw_w[idx];
  }
  for (int idx = tid; idx < RBC*IBAND; idx += 256) s_band[idx] = band_w[idx];
  for (int idx = tid; idx < DMODEL*RBC; idx += 256) s_pw[idx] = pw_w[idx];
  __syncthreads();
  if (tid < RBC) {            // sum of depthwise kernel (for band_b constant term)
    float sd = 0.f;
    for (int pq = 0; pq < 256; ++pq) sd += s_dwq[pq*RBC + tid];
    s_sumdw[tid] = sd;
  }

  const int b  = blockIdx.x >> 4;
  const int pi = blockIdx.x & 15;
  const int col = tid;
  const int qq = col & 15;

  // per-thread depthwise weights for this column-in-patch: 16x8 in registers
  float dq[PP][RBC];
  #pragma unroll
  for (int p = 0; p < PP; ++p)
    #pragma unroll
    for (int c = 0; c < RBC; ++c)
      dq[p][c] = s_dwq[(p*16 + qq)*RBC + c];

  float acc[RBC];
  #pragma unroll
  for (int c = 0; c < RBC; ++c) acc[c] = 0.f;

  const float* xbase = x + ((size_t)b*IBAND)*(HDIM*WDIM) + (size_t)pi*PP*WDIM + col;
  for (int ib = 0; ib < IBAND; ++ib) {
    const float* xp = xbase + (size_t)ib*(HDIM*WDIM);
    __builtin_prefetch(xp + HDIM*WDIM, 0, 0);   // global_prefetch_b8, next band
    float y[RBC];
    #pragma unroll
    for (int c = 0; c < RBC; ++c) y[c] = 0.f;
    #pragma unroll
    for (int p = 0; p < PP; ++p) {
      float val = xp[p*WDIM];
      #pragma unroll
      for (int c = 0; c < RBC; ++c) y[c] = fmaf(val, dq[p][c], y[c]);
    }
    #pragma unroll
    for (int c = 0; c < RBC; ++c) acc[c] = fmaf(s_band[c*IBAND + ib], y[c], acc[c]);
  }

  #pragma unroll
  for (int c = 0; c < RBC; ++c) s_red[col*RBC + c] = acc[c];
  __syncthreads();

  if (tid < 16*RBC) {          // reduce 16 columns of each patch
    int jj = tid >> 3, c = tid & 7;
    float s = 0.f;
    for (int qx = 0; qx < 16; ++qx) s += s_red[(jj*16 + qx)*RBC + c];
    s += band_b[c] * s_sumdw[c];
    s_dw[jj*RBC + c] = s;
  }
  __syncthreads();

  // pointwise 8 -> 192
  for (int r = tid; r < 16*DMODEL; r += 256) {
    int jj = r / DMODEL, d = r % DMODEL;
    float s = 0.f;
    #pragma unroll
    for (int c = 0; c < RBC; ++c) s = fmaf(s_pw[d*RBC + c], s_dw[jj*RBC + c], s);
    tok_raw[((size_t)b*NTOK + pi*16 + jj)*DMODEL + d] = s;
  }
}

// ============================================================
// LayerNorm over DMODEL, one block (192 threads) per row.
// ============================================================
__global__ __launch_bounds__(DMODEL) void k_ln(
    const float* __restrict__ in, const float* __restrict__ w,
    const float* __restrict__ bsh, float* __restrict__ outf,
    __bf16* __restrict__ outbf)
{
  const int row = blockIdx.x, t = threadIdx.x;
  __shared__ float s[DMODEL];
  float v = in[(size_t)row*DMODEL + t];
  s[t] = v; __syncthreads();
  for (int off = 96; off >= 3; off >>= 1) { if (t < off) s[t] += s[t+off]; __syncthreads(); }
  float mean = (s[0] + s[1] + s[2]) * (1.f/DMODEL);
  float d0 = v - mean;
  __syncthreads();
  s[t] = d0*d0; __syncthreads();
  for (int off = 96; off >= 3; off >>= 1) { if (t < off) s[t] += s[t+off]; __syncthreads(); }
  float var = (s[0] + s[1] + s[2]) * (1.f/DMODEL);
  float y = d0 * rsqrtf(var + 1e-5f) * w[t] + bsh[t];
  if (outf)  outf[(size_t)row*DMODEL + t] = y;
  if (outbf) outbf[(size_t)row*DMODEL + t] = (__bf16)y;
}

// row softmax over DMODEL -> bf16
__global__ __launch_bounds__(DMODEL) void k_softmax(
    const float* __restrict__ q, __bf16* __restrict__ out)
{
  const int row = blockIdx.x, t = threadIdx.x;
  __shared__ float s[DMODEL];
  float v = q[(size_t)row*DMODEL + t];
  s[t] = v; __syncthreads();
  for (int off = 96; off >= 3; off >>= 1) { if (t < off) s[t] = fmaxf(s[t], s[t+off]); __syncthreads(); }
  float mx = fmaxf(fmaxf(s[0], s[1]), s[2]);
  __syncthreads();
  float e = expf(v - mx);
  s[t] = e; __syncthreads();
  for (int off = 96; off >= 3; off >>= 1) { if (t < off) s[t] += s[t+off]; __syncthreads(); }
  float sum = s[0] + s[1] + s[2];
  out[(size_t)row*DMODEL + t] = (__bf16)(e / sum);
}

// ============================================================
// Router: top-2 over spec/spat logits, tau-softmax gate, prototype mix.
// Writes bf16 row [spec(8) | spat(192) | zero-pad to 224] for the keys GEMM.
// ============================================================
__global__ void k_router(const float* __restrict__ rlog,
                         const float* __restrict__ spec_proto,
                         const float* __restrict__ spat_proto,
                         __bf16* __restrict__ mixed)
{
  int tkn = blockIdx.x*blockDim.x + threadIdx.x;
  if (tkn >= NROWS) return;
  const float* rl = rlog + (size_t)tkn*48;
  int i1 = 0; float v1 = rl[0];
  for (int k = 1; k < KS; ++k) { float tv = rl[k]; if (tv > v1) { v1 = tv; i1 = k; } }
  int i2 = (i1 == 0) ? 1 : 0; float v2 = rl[i2];
  for (int k = 0; k < KS; ++k) { if (k == i1) continue; float tv = rl[k]; if (tv > v2) { v2 = tv; i2 = k; } }
  float e2 = expf((v2 - v1) / TAU);
  float a1 = 1.f / (1.f + e2), a2 = e2 / (1.f + e2);
  int p1 = 0; float u1 = rl[KS];
  for (int k = 1; k < KS; ++k) { float tv = rl[KS+k]; if (tv > u1) { u1 = tv; p1 = k; } }
  int p2 = (p1 == 0) ? 1 : 0; float u2 = rl[KS+p2];
  for (int k = 0; k < KS; ++k) { if (k == p1) continue; float tv = rl[KS+k]; if (tv > u2) { u2 = tv; p2 = k; } }
  float f2 = expf((u2 - u1) / TAU);
  float b1 = 1.f / (1.f + f2), b2 = f2 / (1.f + f2);

  __bf16* mrow = mixed + (size_t)tkn*224;
  #pragma unroll
  for (int c = 0; c < RBC; ++c)
    mrow[c] = (__bf16)(a1*spec_proto[i1*RBC + c] + a2*spec_proto[i2*RBC + c]);
  for (int d = 0; d < DMODEL; ++d)
    mrow[RBC + d] = (__bf16)(b1*spat_proto[p1*DMODEL + d] + b2*spat_proto[p2*DMODEL + d]);
  for (int d = 200; d < 224; ++d) mrow[d] = (__bf16)0.f;
}

// weight repack: f32 (R,C) -> bf16 group-of-8 K-packed ((Rpad/8), C, 8), zero padded
__global__ void k_pack_w(const float* __restrict__ W, __bf16* __restrict__ out,
                         int R, int Rpad, int C)
{
  int idx = blockIdx.x*256 + threadIdx.x;
  if (idx >= Rpad*C) return;
  int k = idx / C, n = idx % C;
  float v = (k < R) ? W[(size_t)k*C + n] : 0.f;
  out[(size_t)(k >> 3)*(8*C) + n*8 + (k & 7)] = (__bf16)v;
}

// ============================================================
// Generic batched bf16 WMMA GEMM: C = act(A@B + bias) * scale [(+=)]
// A: bf16 row-major (lda = K stride), B: bf16 group-of-8 K-packed.
// One wave -> one 16x16 tile, 8 waves per block, K in steps of 32.
// Fragments: 2x global_load_b128 per matrix per step, pointer-increment only.
// ============================================================
__global__ __launch_bounds__(256) void k_gemm_bf16(
    const __bf16* __restrict__ A, const __bf16* __restrict__ Bp,
    void* __restrict__ Cout, const float* __restrict__ bias,
    const float* __restrict__ scale,
    int M, int N, int K, int NB,
    long sA, long sB, long sC, int lda, int ldc,
    int mode, int gelu_flag)
{
  const int lane = threadIdx.x & 31;
  const int wave = threadIdx.x >> 5;
  const int MT = (M + 15) >> 4, NT = N >> 4;
  const int tile = blockIdx.x*8 + wave;
  if (tile >= NB*MT*NT) return;           // wave-uniform: EXEC stays all-ones
  const int batch = tile / (MT*NT);
  const int rem = tile % (MT*NT);
  const int mt = rem / NT, nt = rem % NT;
  const int ml = lane & 15;
  const int hi = lane >> 4;               // 0 for lanes 0..15, 1 for 16..31
  const int khalf = hi << 3;              // +8 K offset for lanes 16..31

  const size_t bstep = (size_t)8*N;       // one k-group of 8 rows
  const __bf16* Acur = A + (size_t)batch*sA + (size_t)(mt*16 + ml)*lda + khalf;
  const __bf16* Bcur = Bp + (size_t)batch*sB + (size_t)(nt*16 + ml)*8
                          + (hi ? bstep : 0);

  v8f c = {0.f,0.f,0.f,0.f,0.f,0.f,0.f,0.f};
  #pragma unroll 2
  for (int kk = 0; kk < K; kk += 32) {
    v8bf a_lo = *(const v8bf*)(Acur);
    v8bf a_hi = *(const v8bf*)(Acur + 16);
    v8bf b_lo = *(const v8bf*)(Bcur);
    v8bf b_hi = *(const v8bf*)(Bcur + 2*bstep);
    Acur += 32;
    Bcur += 4*bstep;
    v16bf a, b;
    #pragma unroll
    for (int e = 0; e < 8; ++e) {
      a[e] = a_lo[e]; a[8+e] = a_hi[e];
      b[e] = b_lo[e]; b[8+e] = b_hi[e];
    }
    c = __builtin_amdgcn_wmma_f32_16x16x32_bf16(false, a, false, b, (short)0, c,
                                                false, false);
  }

  const int n_g = nt*16 + ml;
  const float bn = bias ? bias[n_g] : 0.f;
  const float sc = scale ? scale[n_g] : 1.f;
  #pragma unroll
  for (int r = 0; r < 8; ++r) {
    int m_g = mt*16 + r + (hi << 3);
    if (m_g >= M) continue;
    float val = c[r] + bn;
    if (gelu_flag) val = 0.5f*val*(1.f + erff(val*0.70710678118654752f));
    val *= sc;
    if (mode == OUT_F32)
      ((float*)Cout)[(size_t)batch*sC + (size_t)m_g*ldc + n_g] = val;
    else if (mode == OUT_F32_RESID)
      ((float*)Cout)[(size_t)batch*sC + (size_t)m_g*ldc + n_g] += val;
    else if (mode == OUT_BF_RM)
      ((__bf16*)Cout)[(size_t)batch*sC + (size_t)m_g*ldc + n_g] = (__bf16)val;
    else if (mode == OUT_BF_PP)
      ((__bf16*)Cout)[(size_t)batch*sC + (size_t)(m_g >> 3)*(8*ldc) + n_g*8 + (m_g & 7)] = (__bf16)val;
    else  // OUT_BF_TR
      ((__bf16*)Cout)[(size_t)batch*sC + (size_t)n_g*ldc + m_g] = (__bf16)val;
  }
}

// ============================================================
// Final: mean over tokens -> LN -> head (192x9). One block per batch.
// ============================================================
__global__ __launch_bounds__(DMODEL) void k_head(
    const float* __restrict__ xb, const float* __restrict__ flw,
    const float* __restrict__ flb, const float* __restrict__ hw,
    const float* __restrict__ hb, float* __restrict__ out)
{
  const int b = blockIdx.x, t = threadIdx.x;
  __shared__ float s[DMODEL];
  float sum = 0.f;
  const float* base = xb + (size_t)b*NTOK*DMODEL + t;
  for (int n = 0; n < NTOK; ++n) sum += base[(size_t)n*DMODEL];
  float m = sum * (1.f/NTOK);
  s[t] = m; __syncthreads();
  for (int off = 96; off >= 3; off >>= 1) { if (t < off) s[t] += s[t+off]; __syncthreads(); }
  float mean = (s[0] + s[1] + s[2]) * (1.f/DMODEL);
  float d0 = m - mean;
  __syncthreads();
  s[t] = d0*d0; __syncthreads();
  for (int off = 96; off >= 3; off >>= 1) { if (t < off) s[t] += s[t+off]; __syncthreads(); }
  float var = (s[0] + s[1] + s[2]) * (1.f/DMODEL);
  float y = d0 * rsqrtf(var + 1e-5f) * flw[t] + flb[t];
  __syncthreads();
  s[t] = y; __syncthreads();
  if (t < NC) {
    float o = hb[t];
    for (int d = 0; d < DMODEL; ++d) o = fmaf(s[d], hw[d*NC + t], o);
    out[b*NC + t] = o;
  }
}

// ============================================================
static inline void gemm(hipStream_t st, const __bf16* A, const __bf16* B, void* C,
                        const float* bias, const float* scale,
                        int M, int N, int K, int NB, long sA, long sB, long sC,
                        int lda, int ldc, int mode, int gelu)
{
  int tiles = NB * (M/16) * (N/16);
  int blocks = (tiles + 7)/8;
  k_gemm_bf16<<<blocks, 256, 0, st>>>(A, B, C, bias, scale, M, N, K, NB,
                                      sA, sB, sC, lda, ldc, mode, gelu);
}

extern "C" void kernel_launch(void* const* d_in, const int* in_sizes, int n_in,
                              void* d_out, int out_size, void* d_ws, size_t ws_size,
                              hipStream_t stream)
{
  const float* x          = (const float*)d_in[0];
  const float* band_w     = (const float*)d_in[1];
  const float* band_b     = (const float*)d_in[2];
  const float* dw_w       = (const float*)d_in[3];
  const float* pw_w       = (const float*)d_in[4];
  const float* pe_ln_w    = (const float*)d_in[5];
  const float* pe_ln_b    = (const float*)d_in[6];
  const float* spec_proto = (const float*)d_in[7];
  const float* spat_proto = (const float*)d_in[8];
  const float* r1_w       = (const float*)d_in[9];
  const float* r1_b       = (const float*)d_in[10];
  const float* r2_w       = (const float*)d_in[11];
  const float* r2_b       = (const float*)d_in[12];
  const float* key_w      = (const float*)d_in[13];
  const float* pos_bias   = (const float*)d_in[14];
  const float* blk_ln_w   = (const float*)d_in[15];
  const float* blk_ln_b   = (const float*)d_in[16];
  const float* blk_q_w    = (const float*)d_in[17];
  const float* blk_v_w    = (const float*)d_in[18];
  const float* blk_f1_w   = (const float*)d_in[19];
  const float* blk_f1_b   = (const float*)d_in[20];
  const float* blk_f2_w   = (const float*)d_in[21];
  const float* blk_f2_b   = (const float*)d_in[22];
  const float* blk_g1     = (const float*)d_in[23];
  const float* blk_g2     = (const float*)d_in[24];
  const float* fin_ln_w   = (const float*)d_in[25];
  const float* fin_ln_b   = (const float*)d_in[26];
  const float* head_w     = (const float*)d_in[27];
  const float* head_b     = (const float*)d_in[28];
  float* out = (float*)d_out;

  // ---- workspace carve (all offsets 256B aligned) ----
  unsigned char* p = (unsigned char*)d_ws;
  auto carve = [&](size_t bytes) -> void* {
    void* r = (void*)p; p += (bytes + 255) & ~(size_t)255; return r;
  };
  float*  tok_raw = (float*)carve((size_t)NROWS*DMODEL*4);   // reused as q
  float*  xb      = (float*)carve((size_t)NROWS*DMODEL*4);
  __bf16* tok_bf  = (__bf16*)carve((size_t)NROWS*DMODEL*2);
  __bf16* h_bf    = (__bf16*)carve((size_t)NROWS*128*2);
  float*  rlog    = (float*)carve((size_t)NROWS*48*4);
  __bf16* mixed   = (__bf16*)carve((size_t)NROWS*224*2);
  __bf16* keysT   = (__bf16*)carve((size_t)BB*DMODEL*NTOK*2);
  __bf16* xn_bf   = (__bf16*)carve((size_t)NROWS*DMODEL*2);
  __bf16* qs_bf   = (__bf16*)carve((size_t)NROWS*DMODEL*2);
  __bf16* vp      = (__bf16*)carve((size_t)NROWS*DMODEL*2);   // group-of-8 packed
  __bf16* KVp     = (__bf16*)carve((size_t)BB*DMODEL*DMODEL*2);
  __bf16* ff_bf   = (__bf16*)carve((size_t)NROWS*FFH*2);
  __bf16* r1p     = (__bf16*)carve((size_t)DMODEL*128*2);
  __bf16* r2p     = (__bf16*)carve((size_t)128*48*2);
  __bf16* kwp     = (__bf16*)carve((size_t)224*DMODEL*2);
  __bf16* qp      = (__bf16*)carve((size_t)3*DMODEL*DMODEL*2);
  __bf16* vpw     = (__bf16*)carve((size_t)3*DMODEL*DMODEL*2);
  __bf16* f1p     = (__bf16*)carve((size_t)3*DMODEL*FFH*2);
  __bf16* f2p     = (__bf16*)carve((size_t)3*FFH*DMODEL*2);
  float*  qf      = tok_raw;   // reuse tok_raw after pe-LN

  // ---- weight repacks (bf16 group-of-8 K-packed) ----
  auto pack = [&](const float* W, __bf16* o, int R, int Rpad, int C) {
    int tot = Rpad*C;
    k_pack_w<<<(tot + 255)/256, 256, 0, stream>>>(W, o, R, Rpad, C);
  };
  pack(r1_w, r1p, DMODEL, DMODEL, 128);
  pack(r2_w, r2p, 128, 128, 48);
  pack(key_w, kwp, 200, 224, DMODEL);
  for (int i = 0; i < 3; ++i) {
    pack(blk_q_w  + (size_t)i*DMODEL*DMODEL, qp  + (size_t)i*DMODEL*DMODEL, DMODEL, DMODEL, DMODEL);
    pack(blk_v_w  + (size_t)i*DMODEL*DMODEL, vpw + (size_t)i*DMODEL*DMODEL, DMODEL, DMODEL, DMODEL);
    pack(blk_f1_w + (size_t)i*DMODEL*FFH,    f1p + (size_t)i*DMODEL*FFH,    DMODEL, DMODEL, FFH);
    pack(blk_f2_w + (size_t)i*FFH*DMODEL,    f2p + (size_t)i*FFH*DMODEL,    FFH,    FFH,    DMODEL);
  }

  // ---- 1) fused patch embed (memory bound, reads x once) ----
  k_patch_embed<<<BB*16, 256, 0, stream>>>(x, band_w, band_b, dw_w, pw_w, tok_raw);

  // ---- 2) pe-LN -> xb (f32 residual) + tok_bf (WMMA A) ----
  k_ln<<<NROWS, DMODEL, 0, stream>>>(tok_raw, pe_ln_w, pe_ln_b, xb, tok_bf);

  // ---- 3) router MLP (WMMA) ----
  gemm(stream, tok_bf, r1p, h_bf, r1_b, nullptr,
       NROWS, 128, DMODEL, 1, 0, 0, 0, DMODEL, 128, OUT_BF_RM, 1);
  gemm(stream, h_bf, r2p, rlog, r2_b, nullptr,
       NROWS, 48, 128, 1, 0, 0, 0, 128, 48, OUT_F32, 0);

  // ---- 4) top-2 gating + prototype mix ----
  k_router<<<(NROWS + 255)/256, 256, 0, stream>>>(rlog, spec_proto, spat_proto, mixed);

  // ---- 5) keys = mixed @ key_w + pos_bias, stored transposed bf16 per batch ----
  gemm(stream, mixed, kwp, keysT, pos_bias, nullptr,
       NTOK, DMODEL, 224, BB, (long)NTOK*224, 0, (long)DMODEL*NTOK,
       224, NTOK, OUT_BF_TR, 0);

  // ---- 6) linear-attention blocks ----
  for (int i = 0; i < 3; ++i) {
    const float* lw = blk_ln_w + (size_t)i*DMODEL;
    const float* lb = blk_ln_b + (size_t)i*DMODEL;
    k_ln<<<NROWS, DMODEL, 0, stream>>>(xb, lw, lb, nullptr, xn_bf);
    // q, v
    gemm(stream, xn_bf, qp + (size_t)i*DMODEL*DMODEL, qf, nullptr, nullptr,
         NROWS, DMODEL, DMODEL, 1, 0, 0, 0, DMODEL, DMODEL, OUT_F32, 0);
    gemm(stream, xn_bf, vpw + (size_t)i*DMODEL*DMODEL, vp, nullptr, nullptr,
         NROWS, DMODEL, DMODEL, 1, 0, 0, 0, DMODEL, DMODEL, OUT_BF_PP, 0);
    k_softmax<<<NROWS, DMODEL, 0, stream>>>(qf, qs_bf);
    // KV[b] = keys^T @ v   (batched, K = 256 tokens)
    gemm(stream, keysT, vp, KVp, nullptr, nullptr,
         DMODEL, DMODEL, NTOK, BB, (long)DMODEL*NTOK, (long)NTOK*DMODEL,
         (long)DMODEL*DMODEL, NTOK, DMODEL, OUT_BF_PP, 0);
    // xb += (q_s @ KV) * g1
    gemm(stream, qs_bf, KVp, xb, nullptr, blk_g1 + (size_t)i*DMODEL,
         NTOK, DMODEL, DMODEL, BB, (long)NTOK*DMODEL, (long)DMODEL*DMODEL,
         (long)NTOK*DMODEL, DMODEL, DMODEL, OUT_F32_RESID, 0);
    // FFN
    k_ln<<<NROWS, DMODEL, 0, stream>>>(xb, lw, lb, nullptr, xn_bf);
    gemm(stream, xn_bf, f1p + (size_t)i*DMODEL*FFH, ff_bf,
         blk_f1_b + (size_t)i*FFH, nullptr,
         NROWS, FFH, DMODEL, 1, 0, 0, 0, DMODEL, FFH, OUT_BF_RM, 1);
    gemm(stream, ff_bf, f2p + (size_t)i*FFH*DMODEL, xb,
         blk_f2_b + (size_t)i*DMODEL, blk_g2 + (size_t)i*DMODEL,
         NROWS, DMODEL, FFH, 1, 0, 0, 0, FFH, DMODEL, OUT_F32_RESID, 0);
  }

  // ---- 7) mean pool + final LN + head ----
  k_head<<<BB, DMODEL, 0, stream>>>(xb, fin_ln_w, fin_ln_b, head_w, head_b, out);

  (void)in_sizes; (void)n_in; (void)out_size; (void)ws_size;
}
